// GAT_54065048323042
// MI455X (gfx1250) — matrix operand order repaired and checked by reading, not compiled, not run on hardware
//
#include <hip/hip_runtime.h>
#include <hip/hip_bf16.h>
#include <math.h>

#define NN     50000
#define EE     800000
#define INCH   128
#define HID    16
#define HEADS  8
#define OUTCH  64
#define NSLOPE 0.2f

typedef __attribute__((ext_vector_type(16))) _Float16 v16h;
typedef __attribute__((ext_vector_type(8)))  _Float16 v8h;
typedef __attribute__((ext_vector_type(8)))  float    v8f;

union V16U { v16h v; v8h h[2]; };

// ---------------------------------------------------------------------------
// f32 -> f16 conversion
__global__ void gat_cvt_f16(const float* __restrict__ src, _Float16* __restrict__ dst, int n) {
    int i = blockIdx.x * blockDim.x + threadIdx.x;
    if (i < n) dst[i] = (_Float16)src[i];
}

// Pack W [K x NC] (row-major f32) into WMMA B-operand lane layout (f16).
// Output index = (((kt*ntiles + nt)*32 + lane)*16 + j)
//   lane<16: K = kt*32 + j      (j=0..15), N = nt*16 + lane
//   lane>=16: K = kt*32 + 16 + j,          N = nt*16 + (lane-16)
__global__ void gat_pack_w(const float* __restrict__ W, _Float16* __restrict__ Wp, int K, int NC) {
    int idx = blockIdx.x * blockDim.x + threadIdx.x;
    if (idx >= K * NC) return;
    int j    = idx & 15;
    int lane = (idx >> 4) & 31;
    int tile = idx >> 9;
    int ntiles = NC >> 4;
    int nt = tile % ntiles;
    int kt = tile / ntiles;
    int k = kt * 32 + ((lane < 16) ? 0 : 16) + j;
    int n = nt * 16 + (lane & 15);
    Wp[idx] = (_Float16)W[k * NC + n];
}

// ---------------------------------------------------------------------------
// C[M x NC] = A[M x K] * B[K x NC], A f16 row-major, B pre-packed, C f32.
// One wave per 16x16 tile. block = (32,4): 4 N-tiles per block.
__global__ __launch_bounds__(128)
void gat_wmma_gemm(const _Float16* __restrict__ A, const _Float16* __restrict__ Bp,
                   float* __restrict__ C, int K, int NC) {
    int lane = threadIdx.x;                        // 0..31
    int mt   = blockIdx.x;
    int nt   = blockIdx.y * 4 + threadIdx.y;       // ntiles is a multiple of 4
    int ntiles = NC >> 4;
    int ktiles = K >> 5;

    int m    = mt * 16 + (lane & 15);
    int koff = (lane < 16) ? 0 : 16;               // A layout half-wave K offset

    v8f acc = {};
    const _Float16* arow = A + (size_t)m * K;
    for (int kt = 0; kt < ktiles; ++kt) {
        V16U a;
        const _Float16* ap = arow + kt * 32 + koff;
        a.h[0] = *(const v8h*)(ap);                // K = kt*32 + koff .. +7
        a.h[1] = *(const v8h*)(ap + 8);            // K = kt*32 + koff + 8 .. +15
        v16h b = *(const v16h*)(Bp + (((size_t)kt * ntiles + nt) * 32 + lane) * 16);
        acc = __builtin_amdgcn_wmma_f32_16x16x32_f16(
            false, a.v, false, b, (short)0, acc, false, false);
    }
    // C layout: VGPR r, lanes 0-15 -> M=r, lanes 16-31 -> M=8+r; N = lane&15
    int row0 = mt * 16 + (lane >> 4) * 8;
    int col  = nt * 16 + (lane & 15);
#pragma unroll
    for (int r = 0; r < 8; ++r)
        C[(size_t)(row0 + r) * NC + col] = acc[r];
}

// ---------------------------------------------------------------------------
// s[n,h] = dot(h[n,h,:], a_src[h,:]); d likewise.
__global__ void gat_attn_proj(const float* __restrict__ hfeat,
                              const float* __restrict__ a_src,
                              const float* __restrict__ a_dst,
                              float* __restrict__ s, float* __restrict__ d,
                              int n_nodes, int heads, int ch) {
    int t = blockIdx.x * blockDim.x + threadIdx.x;
    if (t >= n_nodes * heads) return;
    int hd = t % heads;
    int n  = t / heads;
    const float* hp = hfeat + (size_t)n * heads * ch + hd * ch;
    const float* as = a_src + hd * ch;
    const float* ad = a_dst + hd * ch;
    float ss = 0.f, dd = 0.f;
    for (int c = 0; c < ch; ++c) { float v = hp[c]; ss += v * as[c]; dd += v * ad[c]; }
    s[t] = ss; d[t] = dd;
}

__global__ void gat_init_mz(float* __restrict__ m, float* __restrict__ z, int n) {
    int i = blockIdx.x * blockDim.x + threadIdx.x;
    if (i < n) { m[i] = -INFINITY; z[i] = 0.f; }
}

__global__ void gat_init_bias(float* __restrict__ agg, const float* __restrict__ bias,
                              int n_nodes, int feat) {
    int i = blockIdx.x * blockDim.x + threadIdx.x;
    if (i < n_nodes * feat) agg[i] = bias[i % feat];
}

// ---------------------------------------------------------------------------
__device__ __forceinline__ void atomicMaxF(float* addr, float v) {
    // works for all orderings given init = -inf
    if (v >= 0.f) atomicMax((int*)addr, __float_as_int(v));
    else          atomicMin((unsigned int*)addr, __float_as_uint(v));
}

__device__ __forceinline__ float lrelu(float v) { return v > 0.f ? v : NSLOPE * v; }

__global__ void gat_edge_max(const long long* __restrict__ ei,
                             const float* __restrict__ s, const float* __restrict__ d,
                             float* __restrict__ m, int ne, int H) {
    int t = blockIdx.x * blockDim.x + threadIdx.x;
    if (t >= ne * H) return;
    int h = t % H;
    int e = t / H;
    int sr = (int)ei[e];
    int ds = (int)ei[ne + e];
    float v = lrelu(s[sr * H + h] + d[ds * H + h]);
    atomicMaxF(&m[ds * H + h], v);
}

__global__ void gat_edge_expsum(const long long* __restrict__ ei,
                                const float* __restrict__ s, const float* __restrict__ d,
                                const float* __restrict__ m, float* __restrict__ z,
                                int ne, int H) {
    int t = blockIdx.x * blockDim.x + threadIdx.x;
    if (t >= ne * H) return;
    int h = t % H;
    int e = t / H;
    int sr = (int)ei[e];
    int ds = (int)ei[ne + e];
    float v  = lrelu(s[sr * H + h] + d[ds * H + h]);
    float mm = m[ds * H + h];
    if (mm < -1e30f) mm = 0.f;                      // isolated-node guard
    atomicAdd(&z[ds * H + h], expf(v - mm));
}

// EPB edges per block; block = EPB*C threads. agg[dst,c] += h[src,c]*alpha[e, c/(C/H)]
template <int H, int C, int EPB>
__global__ void gat_edge_agg(const long long* __restrict__ ei,
                             const float* __restrict__ s, const float* __restrict__ d,
                             const float* __restrict__ m, const float* __restrict__ z,
                             const float* __restrict__ hfeat, float* __restrict__ agg,
                             int ne) {
    __shared__ float sal[EPB * H];
    int tid = threadIdx.x;
    int e0  = blockIdx.x * EPB;                     // grid sized so e0+EPB <= ne
    if (tid < EPB * H) {
        int ae = tid / H, ah = tid % H;
        int ee = e0 + ae;
        int sr = (int)ei[ee];
        int ds = (int)ei[ne + ee];
        float v  = lrelu(s[sr * H + ah] + d[ds * H + ah]);
        float mm = m[ds * H + ah];
        if (mm < -1e30f) mm = 0.f;
        float w = expf(v - mm);
        sal[tid] = w / (z[ds * H + ah] + 1e-16f);
    }
    __syncthreads();
    int le = tid / C, c = tid % C;
    int e  = e0 + le;
    int sr = (int)ei[e];
    int ds = (int)ei[ne + e];
    float val = hfeat[(size_t)sr * C + c] * sal[le * H + c / (C / H)];
    atomicAdd(&agg[(size_t)ds * C + c], val);
}

// ELU then convert to f16 (input to layer-2 GEMM). Bias already folded via init.
__global__ void gat_elu_cvt(const float* __restrict__ agg, _Float16* __restrict__ out, int n) {
    int i = blockIdx.x * blockDim.x + threadIdx.x;
    if (i >= n) return;
    float v = agg[i];
    v = v > 0.f ? v : expm1f(v);
    out[i] = (_Float16)v;
}

// ---------------------------------------------------------------------------
extern "C" void kernel_launch(void* const* d_in, const int* in_sizes, int n_in,
                              void* d_out, int out_size, void* d_ws, size_t ws_size,
                              hipStream_t stream) {
    const float*     x      = (const float*)d_in[0];
    const long long* ei     = (const long long*)d_in[1];   // [2, E] int64
    const float*     W1     = (const float*)d_in[2];
    const float*     a_src1 = (const float*)d_in[3];
    const float*     a_dst1 = (const float*)d_in[4];
    const float*     b1     = (const float*)d_in[5];
    const float*     W2     = (const float*)d_in[6];
    const float*     a_src2 = (const float*)d_in[7];
    const float*     a_dst2 = (const float*)d_in[8];
    const float*     b2     = (const float*)d_in[9];
    float*           out    = (float*)d_out;               // [N, 64]

    // workspace layout (aliased where lifetimes allow)
    char* ws = (char*)d_ws;
    size_t o = 0;
    auto take = [&](size_t bytes) { size_t cur = o; o += (bytes + 255) & ~(size_t)255; return cur; };
    _Float16* xh   = (_Float16*)(ws + take((size_t)NN * INCH * 2));  // x f16; reused as h_elu f16
    _Float16* W1p  = (_Float16*)(ws + take((size_t)INCH * (HEADS * HID) * 2));
    _Float16* W2p  = (_Float16*)(ws + take((size_t)(HEADS * HID) * OUTCH * 2));
    float*    h1   = (float*)(ws + take((size_t)NN * INCH * 4));     // reused as h2 [N,64]
    float*    agg1 = (float*)(ws + take((size_t)NN * INCH * 4));
    float*    s1   = (float*)(ws + take((size_t)NN * HEADS * 4));    // reused as s2
    float*    d1   = (float*)(ws + take((size_t)NN * HEADS * 4));    // reused as d2
    float*    m1   = (float*)(ws + take((size_t)NN * HEADS * 4));    // reused as m2
    float*    z1   = (float*)(ws + take((size_t)NN * HEADS * 4));    // reused as z2
    _Float16* heh = xh;
    float *h2 = h1, *s2 = s1, *d2 = d1, *m2 = m1, *z2 = z1;

    const int TPB = 256;
    auto cdiv = [](int a, int b) { return (a + b - 1) / b; };

    // ---- layer 1 ----
    gat_cvt_f16<<<cdiv(NN * INCH, TPB), TPB, 0, stream>>>(x, xh, NN * INCH);
    gat_pack_w<<<cdiv(INCH * 128, TPB), TPB, 0, stream>>>(W1, W1p, INCH, 128);
    gat_pack_w<<<cdiv(128 * OUTCH, TPB), TPB, 0, stream>>>(W2, W2p, 128, OUTCH);

    gat_wmma_gemm<<<dim3(NN / 16, (128 / 16) / 4), dim3(32, 4), 0, stream>>>(xh, W1p, h1, INCH, 128);

    gat_attn_proj<<<cdiv(NN * HEADS, TPB), TPB, 0, stream>>>(h1, a_src1, a_dst1, s1, d1, NN, HEADS, HID);
    gat_init_mz<<<cdiv(NN * HEADS, TPB), TPB, 0, stream>>>(m1, z1, NN * HEADS);
    gat_init_bias<<<cdiv(NN * 128, TPB), TPB, 0, stream>>>(agg1, b1, NN, 128);

    gat_edge_max<<<cdiv(EE * HEADS, TPB), TPB, 0, stream>>>(ei, s1, d1, m1, EE, HEADS);
    gat_edge_expsum<<<cdiv(EE * HEADS, TPB), TPB, 0, stream>>>(ei, s1, d1, m1, z1, EE, HEADS);
    gat_edge_agg<HEADS, 128, 2><<<EE / 2, 2 * 128, 0, stream>>>(ei, s1, d1, m1, z1, h1, agg1, EE);

    gat_elu_cvt<<<cdiv(NN * 128, TPB), TPB, 0, stream>>>(agg1, heh, NN * 128);

    // ---- layer 2 ----
    gat_wmma_gemm<<<dim3(NN / 16, (OUTCH / 16) / 4), dim3(32, 4), 0, stream>>>(heh, W2p, h2, 128, OUTCH);

    gat_attn_proj<<<cdiv(NN, TPB), TPB, 0, stream>>>(h2, a_src2, a_dst2, s2, d2, NN, 1, OUTCH);
    gat_init_mz<<<cdiv(NN, TPB), TPB, 0, stream>>>(m2, z2, NN);
    gat_init_bias<<<cdiv(NN * OUTCH, TPB), TPB, 0, stream>>>(out, b2, NN, OUTCH);

    gat_edge_max<<<cdiv(EE, TPB), TPB, 0, stream>>>(ei, s2, d2, m2, EE, 1);
    gat_edge_expsum<<<cdiv(EE, TPB), TPB, 0, stream>>>(ei, s2, d2, m2, z2, EE, 1);
    gat_edge_agg<1, OUTCH, 4><<<EE / 4, 4 * OUTCH, 0, stream>>>(ei, s2, d2, m2, z2, h2, out, EE);

    (void)in_sizes; (void)n_in; (void)out_size; (void)ws_size;
}
